// RNN_55173149884419
// MI455X (gfx1250) — matrix operand order repaired and checked by reading, not compiled, not run on hardware
//
#include <hip/hip_runtime.h>
#include <hip/hip_bf16.h>

typedef _Float16 v16h __attribute__((ext_vector_type(16)));
typedef _Float16 v8h  __attribute__((ext_vector_type(8)));
typedef _Float16 v4h  __attribute__((ext_vector_type(4)));
typedef float    v8f  __attribute__((ext_vector_type(8)));

#define T_STEPS 32768
#define KIN_P   512     // 500 padded to mult of 32
#define H_REAL  300
#define H_NP    304     // N padded to mult of 16 (x buffers / weight rows)
#define H_KP    320     // H padded to mult of 32 (K dim of layer-1 / head GEMMs)
#define OUT_REAL 100
#define OUT_NP  112     // 100 padded to mult of 16

// ---------------------------------------------------------------------------
// Pack a float matrix [sr,sc] into a zero-padded f16 matrix [dr,dc].
// ---------------------------------------------------------------------------
__global__ void pack_f16_kernel(const float* __restrict__ src, int sr, int sc,
                                _Float16* __restrict__ dst, int dr, int dc) {
    long long i = (long long)blockIdx.x * blockDim.x + threadIdx.x;
    long long total = (long long)dr * dc;
    if (i >= total) return;
    int r = (int)(i / dc);
    int c = (int)(i % dc);
    float v = (r < sr && c < sc) ? src[(long long)r * sc + c] : 0.0f;
    dst[i] = (_Float16)v;
}

// dst[i] = b1[i] (+ b2[i]) for i < n, else 0 ; dst has npad entries
__global__ void bias_pack_kernel(const float* __restrict__ b1,
                                 const float* __restrict__ b2,
                                 float* __restrict__ dst, int n, int npad) {
    int i = blockIdx.x * blockDim.x + threadIdx.x;
    if (i >= npad) return;
    float v = 0.0f;
    if (i < n) { v = b1[i]; if (b2) v += b2[i]; }
    dst[i] = v;
}

// ---------------------------------------------------------------------------
// WMMA GEMM: C[M, ldc](f32) = A[M, lda](f16, row-major) @ W[., ldw](f16)^T + bias
// One wave computes one 16x16 output tile. K = Ktiles*32 (buffers zero-padded).
// A-fragment: lane l (lr=l&15, hi=l>>4): halves 0..7  = A[m0+lr][k0+hi*8   ..+7]
//                                        halves 8..15 = A[m0+lr][k0+hi*8+16..+7]
// B-fragment: lane l: halves 0..7 = W[n0+lr][k0+hi*8 ..+7], 8..15 = W[n0+lr][k0+hi*8+16..+7]   (B = W^T)
// D layout:   lane l holds D[m0+hi*8+j][n0+lr] in c[j], j=0..7
// ---------------------------------------------------------------------------
__global__ void wmma_gemm_kernel(const _Float16* __restrict__ A, int lda,
                                 const _Float16* __restrict__ W, int ldw,
                                 const float* __restrict__ bias,
                                 float* __restrict__ C, int ldc, int nReal,
                                 int tilesN, int totalTiles, int Ktiles) {
    int wave = blockIdx.x * (blockDim.x >> 5) + (threadIdx.x >> 5);
    if (wave >= totalTiles) return;               // uniform per-wave exit
    int lane = threadIdx.x & 31;
    int lr = lane & 15;
    int hi = lane >> 4;
    int mt = wave / tilesN;
    int nt = wave - mt * tilesN;
    int m0 = mt << 4;
    int n0 = nt << 4;

    const _Float16* arow = A + (size_t)(m0 + lr) * lda + hi * 8;
    const _Float16* brow = W + (size_t)(n0 + lr) * ldw + hi * 8;

    v8f c = {};
    for (int kt = 0; kt < Ktiles; ++kt) {
        int k0 = kt << 5;
        v8h a_lo = *(const v8h*)(arow + k0);
        v8h a_hi = *(const v8h*)(arow + k0 + 16);
        v8h b_lo = *(const v8h*)(brow + k0);
        v8h b_hi = *(const v8h*)(brow + k0 + 16);
        v16h a, b;
#pragma unroll
        for (int i = 0; i < 8; ++i) {
            a[i] = a_lo[i]; a[i + 8] = a_hi[i];
            b[i] = b_lo[i]; b[i + 8] = b_hi[i];
        }
        // (neg_a, A, neg_b, B, c_mod, C, reuse_a, reuse_b)
        c = __builtin_amdgcn_wmma_f32_16x16x32_f16(false, a, false, b,
                                                   (short)0, c, false, false);
    }

    int col = n0 + lr;
    if (col < nReal) {
        float bc = bias[col];
#pragma unroll
        for (int j = 0; j < 8; ++j) {
            C[(size_t)(m0 + hi * 8 + j) * ldc + col] = c[j] + bc;
        }
    }
}

// ---------------------------------------------------------------------------
// Serial recurrence for one RNN layer, single workgroup (320 threads).
// W_hh (300x300 f32) is converted once into LDS f16 (180 KB — needs CDNA5's
// 320 KB/WGP LDS). h lives in LDS f32; reads of h are wave-uniform (LDS
// broadcast). Output written as zero-padded f16 rows [T, 320] for next GEMM.
// ---------------------------------------------------------------------------
__global__ void rnn_recur_kernel(const float* __restrict__ x, int ldx,
                                 const float* __restrict__ Whh,
                                 _Float16* __restrict__ hout, int ldh, int T) {
    extern __shared__ char smem[];
    _Float16* w = (_Float16*)smem;                         // 300*300 f16
    float* h = (float*)(smem + H_REAL * H_REAL * 2);       // 304 f32

    int tid = threadIdx.x;                                 // blockDim = 320
    for (int i = tid; i < H_REAL * H_REAL; i += 320) w[i] = (_Float16)Whh[i];
    if (tid < H_NP) h[tid] = 0.0f;
    __syncthreads();

    const _Float16* wr = w + tid * H_REAL;                 // tid*600 B, 8B aligned
    for (int t = 0; t < T; ++t) {
        float acc = 0.0f;
        if (tid < H_REAL) {
            acc = x[(size_t)t * ldx + tid];
#pragma unroll 4
            for (int k = 0; k < H_REAL; k += 4) {
                v4h wv = *(const v4h*)(wr + k);
                acc += (float)wv[0] * h[k]     + (float)wv[1] * h[k + 1]
                     + (float)wv[2] * h[k + 2] + (float)wv[3] * h[k + 3];
            }
            acc = tanhf(acc);
        }
        __syncthreads();                                   // all h reads done
        if (tid < H_REAL) h[tid] = acc;
        __syncthreads();                                   // h updated for t+1
        hout[(size_t)t * ldh + tid] = (tid < H_REAL) ? (_Float16)acc
                                                     : (_Float16)0.0f;
    }
}

// ---------------------------------------------------------------------------
extern "C" void kernel_launch(void* const* d_in, const int* in_sizes, int n_in,
                              void* d_out, int out_size, void* d_ws, size_t ws_size,
                              hipStream_t stream) {
    const float* input = (const float*)d_in[0];   // [32768, 500]
    const float* W_ih0 = (const float*)d_in[1];   // [300, 500]
    const float* W_hh0 = (const float*)d_in[2];   // [300, 300]
    const float* b_ih0 = (const float*)d_in[3];
    const float* b_hh0 = (const float*)d_in[4];
    const float* W_ih1 = (const float*)d_in[5];   // [300, 300]
    const float* W_hh1 = (const float*)d_in[6];   // [300, 300]
    const float* b_ih1 = (const float*)d_in[7];
    const float* b_hh1 = (const float*)d_in[8];
    const float* W_lin = (const float*)d_in[9];   // [100, 300]
    const float* b_lin = (const float*)d_in[10];
    float* out = (float*)d_out;                   // [32768, 100]

    // -------- workspace layout --------
    char* ws = (char*)d_ws;
    size_t off = 0;
    _Float16* A0f   = (_Float16*)(ws + off); off += (size_t)T_STEPS * KIN_P * 2;  // 32 MB
    _Float16* h0f   = (_Float16*)(ws + off); off += (size_t)T_STEPS * H_KP * 2;   // 20 MB
    _Float16* h1f   = (_Float16*)(ws + off); off += (size_t)T_STEPS * H_KP * 2;   // 20 MB
    float*    xbuf  = (float*)  (ws + off); off += (size_t)T_STEPS * H_NP * 4;    // 40 MB
    _Float16* Wih0f = (_Float16*)(ws + off); off += (size_t)H_NP * KIN_P * 2;
    _Float16* Wih1f = (_Float16*)(ws + off); off += (size_t)H_NP * H_KP * 2;
    _Float16* Wlinf = (_Float16*)(ws + off); off += (size_t)OUT_NP * H_KP * 2;
    float*    bias0 = (float*)  (ws + off); off += H_NP * 4;
    float*    bias1 = (float*)  (ws + off); off += H_NP * 4;
    float*    biasL = (float*)  (ws + off); off += OUT_NP * 4;

    const int PB = 256;
    auto blocks = [](long long n, int b) { return (unsigned)((n + b - 1) / b); };

    // -------- pack f16 operands --------
    pack_f16_kernel<<<blocks((long long)T_STEPS * KIN_P, PB), PB, 0, stream>>>(
        input, T_STEPS, 500, A0f, T_STEPS, KIN_P);
    pack_f16_kernel<<<blocks((long long)H_NP * KIN_P, PB), PB, 0, stream>>>(
        W_ih0, H_REAL, 500, Wih0f, H_NP, KIN_P);
    pack_f16_kernel<<<blocks((long long)H_NP * H_KP, PB), PB, 0, stream>>>(
        W_ih1, H_REAL, H_REAL, Wih1f, H_NP, H_KP);
    pack_f16_kernel<<<blocks((long long)OUT_NP * H_KP, PB), PB, 0, stream>>>(
        W_lin, OUT_REAL, H_REAL, Wlinf, OUT_NP, H_KP);
    bias_pack_kernel<<<blocks(H_NP, 64), 64, 0, stream>>>(b_ih0, b_hh0, bias0, H_REAL, H_NP);
    bias_pack_kernel<<<blocks(H_NP, 64), 64, 0, stream>>>(b_ih1, b_hh1, bias1, H_REAL, H_NP);
    bias_pack_kernel<<<blocks(OUT_NP, 64), 64, 0, stream>>>(b_lin, nullptr, biasL, OUT_REAL, OUT_NP);

    const int tilesM = T_STEPS / 16;   // 2048
    const int wavesPerBlock = 8;       // blockDim 256

    // -------- GEMM 1: x0 = input @ W_ih0^T + (b_ih0+b_hh0) --------
    {
        int tilesN = H_NP / 16;                     // 19
        int total = tilesM * tilesN;
        wmma_gemm_kernel<<<blocks(total, wavesPerBlock), 256, 0, stream>>>(
            A0f, KIN_P, Wih0f, KIN_P, bias0, xbuf, H_NP, H_NP, tilesN, total,
            KIN_P / 32);
    }

    // -------- recurrence layer 0 --------
    size_t shmem = (size_t)H_REAL * H_REAL * 2 + H_NP * 4;   // 181,216 B
    (void)hipFuncSetAttribute((const void*)rnn_recur_kernel,
                              hipFuncAttributeMaxDynamicSharedMemorySize,
                              (int)shmem);
    rnn_recur_kernel<<<1, 320, shmem, stream>>>(xbuf, H_NP, W_hh0, h0f, H_KP, T_STEPS);

    // -------- GEMM 2: x1 = h0 @ W_ih1^T + (b_ih1+b_hh1)  (reuses xbuf) ----
    {
        int tilesN = H_NP / 16;
        int total = tilesM * tilesN;
        wmma_gemm_kernel<<<blocks(total, wavesPerBlock), 256, 0, stream>>>(
            h0f, H_KP, Wih1f, H_KP, bias1, xbuf, H_NP, H_NP, tilesN, total,
            H_KP / 32);
    }

    // -------- recurrence layer 1 --------
    rnn_recur_kernel<<<1, 320, shmem, stream>>>(xbuf, H_NP, W_hh1, h1f, H_KP, T_STEPS);

    // -------- GEMM 3: out = h1 @ W_lin^T + b_lin --------
    {
        int tilesN = OUT_NP / 16;                   // 7
        int total = tilesM * tilesN;
        wmma_gemm_kernel<<<blocks(total, wavesPerBlock), 256, 0, stream>>>(
            h1f, H_KP, Wlinf, H_KP, biasL, out, OUT_REAL, OUT_REAL, tilesN, total,
            H_KP / 32);
    }
}
